// WindowAttention_43173011259815
// MI455X (gfx1250) — matrix operand order repaired and checked by reading, not compiled, not run on hardware
//
#include <hip/hip_runtime.h>
#include <hip/hip_bf16.h>

typedef _Float16 h16;
typedef __attribute__((ext_vector_type(16))) _Float16 v16h;
typedef __attribute__((ext_vector_type(8)))  _Float16 v8h;
typedef __attribute__((ext_vector_type(8)))  float    v8f;
typedef __attribute__((ext_vector_type(4)))  float    v4f;
typedef __attribute__((ext_vector_type(4)))  unsigned int u32x4;
typedef __attribute__((ext_vector_type(8)))  int      i32x8;
typedef __attribute__((ext_vector_type(4)))  int      i32x4;

__device__ __forceinline__ v8f wmma16(v16h a, v16h b, v8f c) {
  return __builtin_amdgcn_wmma_f32_16x16x32_f16(false, a, false, b, (short)0, c, false, false);
}

// TDM: DMA one contiguous 4KB block (512 x 8B elements) from global to LDS.
// D# group0: count=1, lds_addr, global_addr[56:0], type=2.
// D# group1: data_size=3 (8B), tensor_dim0=tile_dim0=512, 1-D tile.
__device__ __forceinline__ void tdm_load_4k(unsigned int lds_addr, const void* gptr) {
  unsigned long long ga = (unsigned long long)(uintptr_t)gptr;
  u32x4 g0;
  g0[0] = 1u;                                     // count = 1
  g0[1] = lds_addr;                               // LDS byte address
  g0[2] = (unsigned int)ga;                       // global_addr[31:0]
  g0[3] = (unsigned int)((ga >> 32) & 0x01FFFFFFu) | 0x80000000u; // addr[56:32] | type=2
  i32x8 g1;
  g1[0] = 0x00030000;                             // workgroup_mask=0, data_size=3 (8B)
  g1[1] = (int)(512u << 16);                      // tensor_dim0 low16 in [31:16]
  g1[2] = (int)(1u << 16);                        // tensor_dim0 hi=0 ; tensor_dim1 lo=1
  g1[3] = (int)(512u << 16);                      // tensor_dim1 hi=0 ; tile_dim0=512
  g1[4] = 0;                                      // tile_dim1=0, tile_dim2=0 (unused)
  g1[5] = 512;                                    // tensor_dim0_stride low32
  g1[6] = 0;
  g1[7] = 0;
  i32x4 z4 = {0, 0, 0, 0};
#if __clang_major__ >= 23
  i32x8 z8 = {0, 0, 0, 0, 0, 0, 0, 0};
  __builtin_amdgcn_tensor_load_to_lds(g0, g1, z4, z4, z8, 0);
#else
  __builtin_amdgcn_tensor_load_to_lds(g0, g1, z4, z4, 0);
#endif
}

// ---------------------------------------------------------------------------
// Workspace layout (bytes):
//   qkv_h  : 3*2048*8*64*32 f16  = 201326592
//   oh_h   : 131072*256    f16   =  67108864
//   wqkv_h : 768*256       f16   =    393216
//   wproj_h: 256*256       f16   =    131072
//   btab   : 8*64*64       f32   =    131072
//   qb     : 768           f32   =      3072
//   hs     : 8             f32   =        32
// ---------------------------------------------------------------------------

// ======================= prep: weights, bias table ========================
__global__ __launch_bounds__(256) void prep_kernel(
    const float* __restrict__ qkv_w, const float* __restrict__ q_bias,
    const float* __restrict__ v_bias, const float* __restrict__ logit_scale,
    const float* __restrict__ w1, const float* __restrict__ b1,
    const float* __restrict__ w2, const float* __restrict__ proj_w,
    h16* __restrict__ wqkv, h16* __restrict__ wproj,
    float* __restrict__ btab, float* __restrict__ qb, float* __restrict__ hs) {
  const int t = threadIdx.x;
  __shared__ float smallTab[225 * 8];
  if (t < 225) {
    int ph = t / 15, pw = t % 15;
    float nh = (float)(ph - 7) * (8.0f / 7.0f);
    float nw = (float)(pw - 7) * (8.0f / 7.0f);
    float th = (nh < 0.f ? -1.f : 1.f) * log2f(fabsf(nh) + 1.f) * (1.0f / 3.0f);
    float tw = (nw < 0.f ? -1.f : 1.f) * log2f(fabsf(nw) + 1.f) * (1.0f / 3.0f);
    float a8[8] = {0.f, 0.f, 0.f, 0.f, 0.f, 0.f, 0.f, 0.f};
    for (int i = 0; i < 512; ++i) {
      float hv = fmaxf(th * w1[2 * i] + tw * w1[2 * i + 1] + b1[i], 0.0f);
#pragma unroll
      for (int hh = 0; hh < 8; ++hh) a8[hh] += hv * w2[hh * 512 + i];
    }
#pragma unroll
    for (int hh = 0; hh < 8; ++hh) smallTab[t * 8 + hh] = a8[hh];
  }
  __syncthreads();
  for (int idx = t; idx < 8 * 64 * 64; idx += 256) {
    int hh = idx >> 12, r = idx & 4095, m = r >> 6, n = r & 63;
    int p = ((m >> 3) - (n >> 3) + 7) * 15 + ((m & 7) - (n & 7) + 7);
    float vv = smallTab[p * 8 + hh];
    btab[idx] = 16.0f / (1.0f + __expf(-vv));
  }
  for (int i = t; i < 768 * 256; i += 256) wqkv[i] = (h16)qkv_w[i];
  for (int i = t; i < 256 * 256; i += 256) wproj[i] = (h16)proj_w[i];
  for (int i = t; i < 768; i += 256) {
    float bv = (i < 256) ? q_bias[i] : ((i < 512) ? 0.0f : v_bias[i - 512]);
    qb[i] = bv;
  }
  if (t < 8) hs[t] = __expf(fminf(logit_scale[t], 4.6051702f));
}

// ============ QKV GEMM: (131072,256) x (256,768) -> scattered f16 =========
__global__ __launch_bounds__(128) void qkv_gemm(
    const float* __restrict__ x, const h16* __restrict__ wq,
    const float* __restrict__ qb, h16* __restrict__ qkv) {
  const int L = threadIdx.x & 31;
  const int wid = threadIdx.x >> 5;
  const int M0 = blockIdx.x * 64 + (wid >> 1) * 32;
  const int N0 = blockIdx.y * 64 + (wid & 1) * 32;
  const int hl = L >> 4, ll = L & 15;

  v8f acc[2][2] = {};
  for (int kb = 0; kb < 256; kb += 32) {
    v16h A[2], B[2];
#pragma unroll
    for (int ti = 0; ti < 2; ++ti) {
      const float* p = x + (size_t)(M0 + ti * 16 + ll) * 256 + kb + hl * 8;
      __builtin_prefetch(p + 32, 0, 0);   // next K slab -> global_prefetch_b8
      v4f f0 = *(const v4f*)p;
      v4f f1 = *(const v4f*)(p + 4);
      v4f f2 = *(const v4f*)(p + 16);
      v4f f3 = *(const v4f*)(p + 20);
      v16h a;
#pragma unroll
      for (int u = 0; u < 4; ++u) {
        a[u] = (h16)f0[u];  a[4 + u]  = (h16)f1[u];
        a[8 + u] = (h16)f2[u]; a[12 + u] = (h16)f3[u];
      }
      A[ti] = a;
    }
#pragma unroll
    for (int tj = 0; tj < 2; ++tj) {
      const h16* p = wq + (size_t)(N0 + tj * 16 + ll) * 256 + kb + hl * 8;
      union { v16h v; v8h h[2]; } u;
      u.h[0] = *(const v8h*)p;
      u.h[1] = *(const v8h*)(p + 16);
      B[tj] = u.v;
    }
#pragma unroll
    for (int ti = 0; ti < 2; ++ti)
#pragma unroll
      for (int tj = 0; tj < 2; ++tj)
        acc[ti][tj] = wmma16(A[ti], B[tj], acc[ti][tj]);
  }

#pragma unroll
  for (int ti = 0; ti < 2; ++ti)
#pragma unroll
    for (int tj = 0; tj < 2; ++tj)
#pragma unroll
      for (int e = 0; e < 8; ++e) {
        int m = M0 + ti * 16 + e + hl * 8;
        int n = N0 + tj * 16 + ll;
        float val = acc[ti][tj][e] + qb[n];
        int b = m >> 6, tok = m & 63;
        int which = n >> 8, rem = n & 255, h = rem >> 5, d = rem & 31;
        size_t idx = ((((size_t)which * 2048 + b) * 8 + h) * 64 + tok) * 32 + d;
        qkv[idx] = (h16)val;
      }
}

// ====================== attention: one wave per (b,h) =====================
// Dynamic LDS layout (21 KB): q@0 (4KB) | k@4096 | v@8192 | P@12288 (8KB)
//                             | qs@20480 (256B) | ks@20736 (256B)
__global__ __launch_bounds__(32) void attn_kernel(
    const h16* __restrict__ qkv, const float* __restrict__ btab,
    const float* __restrict__ hs, h16* __restrict__ oh) {
  extern __shared__ char smem[];
  h16* qL = (h16*)(smem);
  h16* kL = (h16*)(smem + 4096);
  h16* vL = (h16*)(smem + 8192);
  h16* P_s = (h16*)(smem + 12288);
  float* qs_s = (float*)(smem + 20480);
  float* ks_s = (float*)(smem + 20736);

  const int b = blockIdx.x, h = blockIdx.y;
  const int L = threadIdx.x;
  const int hl = L >> 4, ll = L & 15;

  const size_t bh = (size_t)b * 8 + h;
  const h16* qG = qkv + bh * 2048;
  const h16* kG = qkv + (size_t)33554432 + bh * 2048;
  const h16* vG = qkv + (size_t)67108864 + bh * 2048;
  const float sH = hs[h];

  // ---- TDM: stage q,k,v tiles (4KB each, contiguous) into LDS ----
  tdm_load_4k(0u, qG);
  tdm_load_4k(4096u, kG);
  tdm_load_4k(8192u, vG);
  __builtin_amdgcn_s_wait_tensorcnt(0);
  __syncthreads();

  // row norms (2 rows per lane), fold head scale into q's scale
#pragma unroll
  for (int rr = 0; rr < 2; ++rr) {
    int r = 2 * L + rr;
    float ssq = 0.f, ssk = 0.f;
#pragma unroll
    for (int c = 0; c < 4; ++c) {
      v8h qv = *(const v8h*)(qL + r * 32 + c * 8);
      v8h kv = *(const v8h*)(kL + r * 32 + c * 8);
#pragma unroll
      for (int u = 0; u < 8; ++u) {
        float a = (float)qv[u]; ssq += a * a;
        float c2 = (float)kv[u]; ssk += c2 * c2;
      }
    }
    qs_s[r] = sH * rsqrtf(fmaxf(ssq, 1e-24f));
    ks_s[r] = rsqrtf(fmaxf(ssk, 1e-24f));
  }
  __syncthreads();

  // A fragments = normalized q (K = d = 32, exactly one WMMA step)
  v16h Aq[4];
#pragma unroll
  for (int ti = 0; ti < 4; ++ti) {
    int row = ti * 16 + ll;
    float sc = qs_s[row];
    const h16* p = qL + row * 32 + hl * 8;
    v8h c0 = *(const v8h*)p;
    v8h c1 = *(const v8h*)(p + 16);
    v16h a;
#pragma unroll
    for (int u = 0; u < 8; ++u) {
      a[u] = (h16)((float)c0[u] * sc);
      a[8 + u] = (h16)((float)c1[u] * sc);
    }
    Aq[ti] = a;
  }

  v8f acc[4][4] = {};
#pragma unroll
  for (int tj = 0; tj < 4; ++tj) {
    int col = tj * 16 + ll;
    float sc = ks_s[col];
    const h16* p = kL + col * 32 + hl * 8;
    v8h c0 = *(const v8h*)p;
    v8h c1 = *(const v8h*)(p + 16);
    v16h bb;
#pragma unroll
    for (int u = 0; u < 8; ++u) {
      bb[u] = (h16)((float)c0[u] * sc);
      bb[8 + u] = (h16)((float)c1[u] * sc);
    }
#pragma unroll
    for (int ti = 0; ti < 4; ++ti)
      acc[ti][tj] = wmma16(Aq[ti], bb, acc[ti][tj]);
  }

  // + 16*sigmoid(rpb), precomputed table
  const float* bt = btab + h * 4096;
#pragma unroll
  for (int ti = 0; ti < 4; ++ti)
#pragma unroll
    for (int tj = 0; tj < 4; ++tj)
#pragma unroll
      for (int e = 0; e < 8; ++e) {
        int m = ti * 16 + e + hl * 8;
        int n = tj * 16 + ll;
        acc[ti][tj][e] += bt[m * 64 + n];
      }

  // softmax over rows (64 cols spread over 4 tiles x 16 lanes of half-wave)
#pragma unroll
  for (int ti = 0; ti < 4; ++ti)
#pragma unroll
    for (int e = 0; e < 8; ++e) {
      float mx = -1e30f;
#pragma unroll
      for (int tj = 0; tj < 4; ++tj) mx = fmaxf(mx, acc[ti][tj][e]);
#pragma unroll
      for (int d = 1; d < 16; d <<= 1) mx = fmaxf(mx, __shfl_xor(mx, d, 32));
      float sm = 0.f;
#pragma unroll
      for (int tj = 0; tj < 4; ++tj) {
        float p = __expf(acc[ti][tj][e] - mx);
        acc[ti][tj][e] = p;
        sm += p;
      }
#pragma unroll
      for (int d = 1; d < 16; d <<= 1) sm += __shfl_xor(sm, d, 32);
      float inv = 1.0f / sm;
#pragma unroll
      for (int tj = 0; tj < 4; ++tj) acc[ti][tj][e] *= inv;
    }

  // stage P in LDS (C-layout -> A-layout re-shuffle)
#pragma unroll
  for (int ti = 0; ti < 4; ++ti)
#pragma unroll
    for (int tj = 0; tj < 4; ++tj)
#pragma unroll
      for (int e = 0; e < 8; ++e) {
        int m = ti * 16 + e + hl * 8;
        int n = tj * 16 + ll;
        P_s[m * 64 + n] = (h16)acc[ti][tj][e];
      }
  __syncthreads();

  // out = P (64x64) @ V (64x32): two K steps of 32
  v8f o[4][2] = {};
#pragma unroll
  for (int kb2 = 0; kb2 < 2; ++kb2) {
    const int kb = kb2 * 32;
    v16h Ap[4];
#pragma unroll
    for (int ti = 0; ti < 4; ++ti) {
      int row = ti * 16 + ll;
      const h16* p = &P_s[row * 64 + kb + hl * 8];
      union { v16h v; v8h h[2]; } u;
      u.h[0] = *(const v8h*)p;
      u.h[1] = *(const v8h*)(p + 16);
      Ap[ti] = u.v;
    }
    v16h Bv[2];
#pragma unroll
    for (int tj2 = 0; tj2 < 2; ++tj2) {
      int col = tj2 * 16 + ll;
      v16h bb;
#pragma unroll
      for (int u = 0; u < 16; ++u)
        bb[u] = vL[(kb + hl * 8 + u + (u >> 3) * 8) * 32 + col];
      Bv[tj2] = bb;
    }
#pragma unroll
    for (int ti = 0; ti < 4; ++ti)
#pragma unroll
      for (int tj2 = 0; tj2 < 2; ++tj2)
        o[ti][tj2] = wmma16(Ap[ti], Bv[tj2], o[ti][tj2]);
  }

#pragma unroll
  for (int ti = 0; ti < 4; ++ti)
#pragma unroll
    for (int tj2 = 0; tj2 < 2; ++tj2)
#pragma unroll
      for (int e = 0; e < 8; ++e) {
        int m = ti * 16 + e + hl * 8;
        int d = tj2 * 16 + ll;
        oh[((size_t)b * 64 + m) * 256 + h * 32 + d] = (h16)o[ti][tj2][e];
      }
}

// ============= proj GEMM: (131072,256) x (256,256) -> f32 out =============
__global__ __launch_bounds__(128) void proj_gemm(
    const h16* __restrict__ A, const h16* __restrict__ wp,
    const float* __restrict__ pb, float* __restrict__ out) {
  const int L = threadIdx.x & 31;
  const int wid = threadIdx.x >> 5;
  const int M0 = blockIdx.x * 64 + (wid >> 1) * 32;
  const int N0 = blockIdx.y * 64 + (wid & 1) * 32;
  const int hl = L >> 4, ll = L & 15;

  v8f acc[2][2] = {};
  for (int kb = 0; kb < 256; kb += 32) {
    v16h Af[2], Bf[2];
#pragma unroll
    for (int ti = 0; ti < 2; ++ti) {
      const h16* p = A + (size_t)(M0 + ti * 16 + ll) * 256 + kb + hl * 8;
      union { v16h v; v8h h[2]; } u;
      u.h[0] = *(const v8h*)p;
      u.h[1] = *(const v8h*)(p + 16);
      Af[ti] = u.v;
    }
#pragma unroll
    for (int tj = 0; tj < 2; ++tj) {
      const h16* p = wp + (size_t)(N0 + tj * 16 + ll) * 256 + kb + hl * 8;
      union { v16h v; v8h h[2]; } u;
      u.h[0] = *(const v8h*)p;
      u.h[1] = *(const v8h*)(p + 16);
      Bf[tj] = u.v;
    }
#pragma unroll
    for (int ti = 0; ti < 2; ++ti)
#pragma unroll
      for (int tj = 0; tj < 2; ++tj)
        acc[ti][tj] = wmma16(Af[ti], Bf[tj], acc[ti][tj]);
  }

#pragma unroll
  for (int ti = 0; ti < 2; ++ti)
#pragma unroll
    for (int tj = 0; tj < 2; ++tj)
#pragma unroll
      for (int e = 0; e < 8; ++e) {
        int m = M0 + ti * 16 + e + hl * 8;
        int n = N0 + tj * 16 + ll;
        out[(size_t)m * 256 + n] = acc[ti][tj][e] + pb[n];
      }
}

// ===========================================================================
extern "C" void kernel_launch(void* const* d_in, const int* in_sizes, int n_in,
                              void* d_out, int out_size, void* d_ws, size_t ws_size,
                              hipStream_t stream) {
  const float* x           = (const float*)d_in[0];
  const float* qkv_w       = (const float*)d_in[1];
  const float* q_bias      = (const float*)d_in[2];
  const float* v_bias      = (const float*)d_in[3];
  const float* logit_scale = (const float*)d_in[4];
  const float* cpb_w1      = (const float*)d_in[5];
  const float* cpb_b1      = (const float*)d_in[6];
  const float* cpb_w2      = (const float*)d_in[7];
  const float* proj_w      = (const float*)d_in[8];
  const float* proj_b      = (const float*)d_in[9];

  char* w = (char*)d_ws;
  h16*   qkv_h  = (h16*)w;                              // 201326592 B
  h16*   oh_h   = (h16*)(w + 201326592);                //  67108864 B
  h16*   wqkv_h = (h16*)(w + 268435456);                //    393216 B
  h16*   wproj_h= (h16*)(w + 268828672);                //    131072 B
  float* btab   = (float*)(w + 268959744);              //    131072 B
  float* qb     = (float*)(w + 269090816);              //      3072 B
  float* hs     = (float*)(w + 269093888);              //        32 B

  prep_kernel<<<1, 256, 0, stream>>>(qkv_w, q_bias, v_bias, logit_scale,
                                     cpb_w1, cpb_b1, cpb_w2, proj_w,
                                     wqkv_h, wproj_h, btab, qb, hs);
  qkv_gemm<<<dim3(2048, 12), 128, 0, stream>>>(x, wqkv_h, qb, qkv_h);
  attn_kernel<<<dim3(2048, 8), 32, 20992, stream>>>(qkv_h, btab, hs, oh_h);
  proj_gemm<<<dim3(2048, 4), 128, 0, stream>>>(oh_h, wproj_h, proj_b,
                                               (float*)d_out);
}
